// DotProductAttention_72593537237750
// MI455X (gfx1250) — matrix operand order repaired and checked by reading, not compiled, not run on hardware
//
#include <hip/hip_runtime.h>
#include <hip/hip_bf16.h>
#include <math.h>

// ---------------------------------------------------------------------------
// DotProductAttention for MI455X (gfx1250, wave32, WMMA).
//   score  = (Q @ V^T) * (1 - eye)    [diagonal multiplicatively zeroed]
//   attn   = softmax(score, axis=-1)  [written out: 8192x8192 f32 = 256MB]
//   ctx    = attn @ V                 [8192x128 f32]
// Roofline: attn write (256MB) ~= 11us @ 23.3TB/s. Matmuls (2x 17.2GF + PV)
// must run on the bf16 WMMA pipe (16x16x32) to stay near that bound; fp32
// WMMA (16x16x4) would be ~8x slower than the memory floor. So: recompute
// score tiles in bf16, f32 accumulate, two passes (stats, then write+PV).
// ---------------------------------------------------------------------------

#define N_TOK   8192
#define D_DIM   128
#define NWAVES  2
#define THREADS (NWAVES * 32)
#define BM      (NWAVES * 16)   // 32 rows per block (16 per wave)
#define BN      32               // 32 cols per iteration: 2 S-tiles, K=32 for PV

#define SQ_STRIDE (D_DIM + 8)    // bf16 elems; +16B/row to spread LDS banks
#define SV_STRIDE (D_DIM + 8)
#define SP_STRIDE (BN + 2)

typedef __bf16 bf16_t;
typedef bf16_t v16bf __attribute__((ext_vector_type(16)));
typedef float  v8f   __attribute__((ext_vector_type(8)));

// A-matrix 16x32 bf16 element K-index for vector element e, lane half h:
//   VGPR0..3 hold K=(0..7)+8h, VGPR4..7 hold K=(16..23)+8h  (ISA 7.12.2)
__device__ __forceinline__ int a_kidx(int e, int h) {
  return (e < 8 ? e : e + 8) + 8 * h;
}

// Compute one 16x32 S strip: c0 = Q[m0..m0+15] . V[j0..j0+15]^T,
//                            c1 = Q[m0..m0+15] . V[j0+16..j0+31]^T
// sQ: BM x D (bf16, SQ_STRIDE), sV: BN x D (bf16, SV_STRIDE).
__device__ __forceinline__ void compute_S(const bf16_t* __restrict__ sQ,
                                          const bf16_t* __restrict__ sV,
                                          int m0, int lane, v8f& c0, v8f& c1) {
  const int h = lane >> 4;       // lane half
  const int n = lane & 15;       // A-row for this lane / B-column for this lane
  c0 = {};
  c1 = {};
#pragma unroll
  for (int kb = 0; kb < D_DIM; kb += 32) {
    v16bf a, b0, b1;
#pragma unroll
    for (int e = 0; e < 16; ++e) {
      // A: row n of Q tile, lanes 0-15 K=0-15, 16-31 K=16-31.
      a[e]  = sQ[(m0 + n) * SQ_STRIDE + kb + e + 16 * h];
      // B = V^T: B[k][ncol] = V[ncol][k]; lanes 0-15 K=0-15, 16-31 K=16-31.
      b0[e] = sV[(n)      * SV_STRIDE + kb + e + 16 * h];
      b1[e] = sV[(16 + n) * SV_STRIDE + kb + e + 16 * h];
    }
    c0 = __builtin_amdgcn_wmma_f32_16x16x32_bf16(false, a, false, b0,
                                                 (short)0, c0, false, false);
    c1 = __builtin_amdgcn_wmma_f32_16x16x32_bf16(false, a, false, b1,
                                                 (short)0, c1, false, false);
  }
}

// ---------------------------------------------------------------------------
// Pass 1: running row-max / row-sum (log-sum-exp) over all 8192 columns.
// ---------------------------------------------------------------------------
__global__ __launch_bounds__(THREADS)
void attn_pass1_stats(const float* __restrict__ Q, const float* __restrict__ V,
                      float* __restrict__ rowmax, float* __restrict__ rowsum) {
  __shared__ bf16_t sQ[BM * SQ_STRIDE];
  __shared__ bf16_t sV[BN * SV_STRIDE];

  const int tid  = threadIdx.x;
  const int lane = tid & 31;
  const int wave = tid >> 5;
  const int row0 = blockIdx.x * BM;
  const int m0   = wave * 16;
  const int h    = lane >> 4;
  const int n    = lane & 15;

  // Stage Q block as bf16 (coalesced global reads).
  for (int i = tid; i < BM * D_DIM; i += THREADS) {
    const int r = i >> 7, c = i & (D_DIM - 1);
    sQ[r * SQ_STRIDE + c] = (bf16_t)Q[(size_t)(row0 + r) * D_DIM + c];
  }

  float rmax[8], rsum[8];
#pragma unroll
  for (int r = 0; r < 8; ++r) { rmax[r] = -INFINITY; rsum[r] = 0.0f; }

  for (int j0 = 0; j0 < N_TOK; j0 += BN) {
    for (int i = tid; i < BN * D_DIM; i += THREADS) {
      const int r = i >> 7, c = i & (D_DIM - 1);
      sV[r * SV_STRIDE + c] = (bf16_t)V[(size_t)(j0 + r) * D_DIM + c];
    }
    if (j0 + BN < N_TOK)   // global_prefetch_b8 of next V tile (one line/thread)
      __builtin_prefetch((const char*)&V[(size_t)(j0 + BN) * D_DIM] + tid * 128, 0, 0);
    __syncthreads();

    v8f c0, c1;
    compute_S(sQ, sV, m0, lane, c0, c1);

#pragma unroll
    for (int r = 0; r < 8; ++r) {
      const int row_g = row0 + m0 + r + 8 * h;   // C layout: rows r / r+8 by half
      if (row_g == j0 + n)      c0[r] = 0.0f;    // multiplicative diagonal zero
      if (row_g == j0 + 16 + n) c1[r] = 0.0f;

      float tm = fmaxf(c0[r], c1[r]);            // row reduce across 16 lanes
#pragma unroll
      for (int m = 1; m <= 8; m <<= 1) tm = fmaxf(tm, __shfl_xor(tm, m, 32));
      const float nm = fmaxf(rmax[r], tm);
      float s = __expf(c0[r] - nm) + __expf(c1[r] - nm);
#pragma unroll
      for (int m = 1; m <= 8; m <<= 1) s += __shfl_xor(s, m, 32);
      rsum[r] = rsum[r] * __expf(rmax[r] - nm) + s;
      rmax[r] = nm;
    }
    __syncthreads();
  }

  if (n == 0) {
#pragma unroll
    for (int r = 0; r < 8; ++r) {
      const int row_g = row0 + m0 + r + 8 * h;
      rowmax[row_g] = rmax[r];
      rowsum[row_g] = rsum[r];
    }
  }
}

// ---------------------------------------------------------------------------
// Pass 2: recompute S, write attn = exp(S-max)/sum, accumulate ctx = attn @ V.
// ---------------------------------------------------------------------------
__global__ __launch_bounds__(THREADS)
void attn_pass2_write(const float* __restrict__ Q, const float* __restrict__ V,
                      const float* __restrict__ rowmax,
                      const float* __restrict__ rowsum,
                      float* __restrict__ attn, float* __restrict__ ctx) {
  __shared__ bf16_t sQ[BM * SQ_STRIDE];
  __shared__ bf16_t sV[BN * SV_STRIDE];
  __shared__ bf16_t sP[NWAVES * 16 * SP_STRIDE];   // per-wave P re-stripe buffer

  const int tid  = threadIdx.x;
  const int lane = tid & 31;
  const int wave = tid >> 5;
  const int row0 = blockIdx.x * BM;
  const int m0   = wave * 16;
  const int h    = lane >> 4;
  const int n    = lane & 15;
  bf16_t* sp = &sP[wave * 16 * SP_STRIDE];

  for (int i = tid; i < BM * D_DIM; i += THREADS) {
    const int r = i >> 7, c = i & (D_DIM - 1);
    sQ[r * SQ_STRIDE + c] = (bf16_t)Q[(size_t)(row0 + r) * D_DIM + c];
  }

  float rm[8], rsi[8];
#pragma unroll
  for (int r = 0; r < 8; ++r) {
    const int row_g = row0 + m0 + r + 8 * h;
    rm[r]  = rowmax[row_g];
    rsi[r] = 1.0f / rowsum[row_g];
  }

  v8f acc[8];                    // ctx tile: 16 rows x 128 cols (8 WMMA accums)
#pragma unroll
  for (int d = 0; d < 8; ++d) acc[d] = {};

  for (int j0 = 0; j0 < N_TOK; j0 += BN) {
    for (int i = tid; i < BN * D_DIM; i += THREADS) {
      const int r = i >> 7, c = i & (D_DIM - 1);
      sV[r * SV_STRIDE + c] = (bf16_t)V[(size_t)(j0 + r) * D_DIM + c];
    }
    if (j0 + BN < N_TOK)
      __builtin_prefetch((const char*)&V[(size_t)(j0 + BN) * D_DIM] + tid * 128, 0, 0);
    __syncthreads();

    v8f c0, c1;
    compute_S(sQ, sV, m0, lane, c0, c1);

#pragma unroll
    for (int r = 0; r < 8; ++r) {
      const int row_g = row0 + m0 + r + 8 * h;
      if (row_g == j0 + n)      c0[r] = 0.0f;
      if (row_g == j0 + 16 + n) c1[r] = 0.0f;
      const float p0 = __expf(c0[r] - rm[r]) * rsi[r];
      const float p1 = __expf(c1[r] - rm[r]) * rsi[r];
      // attn write: per VGPR row, lanes 0-15 cover 16 consecutive f32 (64B).
      attn[(size_t)row_g * N_TOK + j0 + n]      = p0;
      attn[(size_t)row_g * N_TOK + j0 + 16 + n] = p1;
      // Re-stripe P from C-layout to A-layout via LDS (in-order per wave).
      sp[(r + 8 * h) * SP_STRIDE + n]      = (bf16_t)p0;
      sp[(r + 8 * h) * SP_STRIDE + 16 + n] = (bf16_t)p1;
    }

    // ctx += P(16x32) @ V_tile(32x128): 8 bf16 WMMAs, K=32 matches BN exactly.
    v16bf aP;
#pragma unroll
    for (int e = 0; e < 16; ++e)
      aP[e] = sp[n * SP_STRIDE + e + 16 * h];
#pragma unroll
    for (int d = 0; d < 8; ++d) {
      v16bf bV;
#pragma unroll
      for (int e = 0; e < 16; ++e)
        bV[e] = sV[(e + 16 * h) * SV_STRIDE + d * 16 + n];
      acc[d] = __builtin_amdgcn_wmma_f32_16x16x32_bf16(false, aP, false, bV,
                                                       (short)0, acc[d], false, false);
    }
    __syncthreads();
  }

#pragma unroll
  for (int d = 0; d < 8; ++d)
#pragma unroll
    for (int r = 0; r < 8; ++r) {
      const int row_g = row0 + m0 + r + 8 * h;
      ctx[(size_t)row_g * D_DIM + d * 16 + n] = acc[d][r];
    }
}

// ---------------------------------------------------------------------------
extern "C" void kernel_launch(void* const* d_in, const int* in_sizes, int n_in,
                              void* d_out, int out_size, void* d_ws, size_t ws_size,
                              hipStream_t stream) {
  const float* Q = (const float*)d_in[0];
  const float* V = (const float*)d_in[1];
  float* out  = (float*)d_out;
  float* ctx  = out;                               // [8192,128] first
  float* attn = out + (size_t)N_TOK * D_DIM;       // then [8192,8192]
  float* rowmax = (float*)d_ws;                    // [8192]
  float* rowsum = rowmax + N_TOK;                  // [8192]

  dim3 grid(N_TOK / BM);                           // 256 blocks x 2 waves
  attn_pass1_stats<<<grid, THREADS, 0, stream>>>(Q, V, rowmax, rowsum);
  attn_pass2_write<<<grid, THREADS, 0, stream>>>(Q, V, rowmax, rowsum, attn, ctx);
}
